// SparseAutoencoder_46059229282446
// MI455X (gfx1250) — matrix operand order repaired and checked by reading, not compiled, not run on hardware
//
#include <hip/hip_runtime.h>
#include <hip/hip_bf16.h>

#define B_DIM  8192
#define D_IN   768
#define D_HID  16384
#define TOP_K  30

typedef unsigned short u16;
typedef __attribute__((ext_vector_type(16))) __bf16 v16bf;
typedef __attribute__((ext_vector_type(8)))  float  v8f;

// Types for the async global->LDS DMA builtin:
// param0: AS1 (global) pointer to int4, param1: AS3 (LDS) pointer to int4.
typedef int v4i __attribute__((vector_size(16)));
typedef __attribute__((address_space(1))) v4i g_v4i;
typedef __attribute__((address_space(3))) v4i l_v4i;

#if __has_builtin(__builtin_amdgcn_global_load_async_to_lds_b128) && \
    __has_builtin(__builtin_amdgcn_s_wait_asynccnt)
#define HAVE_ASYNC_LDS 1
#else
#define HAVE_ASYNC_LDS 0
#endif

__device__ __forceinline__ u16 f2bf(float f) {
  union { float f; unsigned u; } c; c.f = f;
  unsigned u = c.u + 0x7FFFu + ((c.u >> 16) & 1u);   // round-to-nearest-even
  return (u16)(u >> 16);
}

// -------------------------------------------------------------------------
// Pre-convert f32 -> bf16 (packed), 8 elements per thread.
// Halves GEMM-side read traffic and enables async DMA staging.
// -------------------------------------------------------------------------
__global__ __launch_bounds__(256) void f32_to_bf16_pack(
    const float* __restrict__ src, u16* __restrict__ dst, int n8) {
  int i = blockIdx.x * blockDim.x + threadIdx.x;
  if (i >= n8) return;
  const float4* s4 = (const float4*)src;
  float4 a = s4[2 * i], b = s4[2 * i + 1];
  uint4 o;
  o.x = (unsigned)f2bf(a.x) | ((unsigned)f2bf(a.y) << 16);
  o.y = (unsigned)f2bf(a.z) | ((unsigned)f2bf(a.w) << 16);
  o.z = (unsigned)f2bf(b.x) | ((unsigned)f2bf(b.y) << 16);
  o.w = (unsigned)f2bf(b.z) | ((unsigned)f2bf(b.w) << 16);
  ((uint4*)dst)[i] = o;
}

// -------------------------------------------------------------------------
// Stage one 128x32 bf16 tile pair into LDS buffer `buf`.
// Async path: GLOBAL_LOAD_ASYNC_TO_LDS_B128, 4 issues/thread (2 A + 2 B).
// -------------------------------------------------------------------------
__device__ __forceinline__ void stage_tiles(
    const u16* __restrict__ xbf, const u16* __restrict__ wbf,
    u16 (*AsmB)[128 * 32], u16 (*BsmB)[128 * 32],
    int buf, int rowBase, int colBase, int k0, int tid) {
#pragma unroll
  for (int i = 0; i < 2; ++i) {
    int q  = tid + i * 256;          // 0..511 chunks of 8 bf16
    int r  = q >> 2;                 // 0..127
    int c8 = (q & 3) << 3;           // 0,8,16,24
    const u16* ga = xbf + (size_t)(rowBase + r) * D_IN + k0 + c8;
    const u16* gb = wbf + (size_t)(colBase + r) * D_IN + k0 + c8;
    u16* la = &AsmB[buf][r * 32 + c8];
    u16* lb = &BsmB[buf][r * 32 + c8];
#if HAVE_ASYNC_LDS
    __builtin_amdgcn_global_load_async_to_lds_b128((g_v4i*)ga, (l_v4i*)la, 0, 0);
    __builtin_amdgcn_global_load_async_to_lds_b128((g_v4i*)gb, (l_v4i*)lb, 0, 0);
#else
    *(uint4*)la = *(const uint4*)ga;
    *(uint4*)lb = *(const uint4*)gb;
#endif
  }
}

// -------------------------------------------------------------------------
// Kernel 1 (fast path): h = relu(xbf @ Wbf^T + b) from pre-converted bf16,
// async double-buffered LDS staging, 8 WMMAs / K-step / wave.
// -------------------------------------------------------------------------
__global__ __launch_bounds__(256, 2) void sae_encoder_wmma_bf16(
    const u16* __restrict__ xbf, const u16* __restrict__ wbf,
    const float* __restrict__ b_enc, float* __restrict__ code_out) {
  __shared__ u16 Asm[2][128 * 32];   // 16 KB double-buffered x tile
  __shared__ u16 Bsm[2][128 * 32];   // 16 KB double-buffered W tile

  const int tid  = threadIdx.x;
  const int lane = tid & 31;
  const int wave = tid >> 5;
  const int wm   = wave & 3;
  const int wn   = wave >> 2;
  const int rowBase = blockIdx.y * 128;
  const int colBase = blockIdx.x * 128;
  const int r16  = lane & 15;
  const int half = lane >> 4;

  v8f acc[2][4];
#pragma unroll
  for (int t = 0; t < 2; ++t)
#pragma unroll
    for (int cn = 0; cn < 4; ++cn)
      acc[t][cn] = (v8f){0.f, 0.f, 0.f, 0.f, 0.f, 0.f, 0.f, 0.f};

  stage_tiles(xbf, wbf, Asm, Bsm, 0, rowBase, colBase, 0, tid);

  int kb = 0;
  for (int k0 = 0; k0 < D_IN; k0 += 32, kb ^= 1) {
    if (k0 + 32 < D_IN) {
      stage_tiles(xbf, wbf, Asm, Bsm, kb ^ 1, rowBase, colBase, k0 + 32, tid);
#if HAVE_ASYNC_LDS
      __builtin_amdgcn_s_wait_asynccnt(4);   // tile-k loads retired (in-order)
#endif
    } else {
#if HAVE_ASYNC_LDS
      __builtin_amdgcn_s_wait_asynccnt(0);
#endif
    }
    __syncthreads();                         // tile kb visible to all waves

    union U { v16bf v; unsigned u[8]; };
    U a[2], b[4];
#pragma unroll
    for (int t = 0; t < 2; ++t) {
      int row = wm * 32 + t * 16 + r16;
      int kbb = half * 8;
#pragma unroll
      for (int v = 0; v < 4; ++v)
        a[t].u[v] = *(const unsigned*)&Asm[kb][row * 32 + kbb + 2 * v];
#pragma unroll
      for (int v = 4; v < 8; ++v)
        a[t].u[v] = *(const unsigned*)&Asm[kb][row * 32 + kbb + 16 + 2 * (v - 4)];
    }
#pragma unroll
    for (int cn = 0; cn < 4; ++cn) {
      int col = wn * 64 + cn * 16 + r16;
#pragma unroll
      for (int g = 0; g < 8; ++g)
        b[cn].u[g] = *(const unsigned*)&Bsm[kb][col * 32 + half * 8 + 2 * g + (g >= 4 ? 8 : 0)];
    }
#pragma unroll
    for (int t = 0; t < 2; ++t)
#pragma unroll
      for (int cn = 0; cn < 4; ++cn)
        acc[t][cn] = __builtin_amdgcn_wmma_f32_16x16x32_bf16(
            false, a[t].v, false, b[cn].v, (short)0, acc[t][cn], false, false);

    __syncthreads();                         // done reading kb before overwrite
  }

#pragma unroll
  for (int t = 0; t < 2; ++t) {
#pragma unroll
    for (int cn = 0; cn < 4; ++cn) {
      int col = colBase + wn * 64 + cn * 16 + r16;
      float bias = b_enc[col];
#pragma unroll
      for (int g = 0; g < 8; ++g) {
        int row = rowBase + wm * 32 + t * 16 + half * 8 + g;
        float v = acc[t][cn][g] + bias;
        v = v > 0.f ? v : 0.f;
        code_out[(size_t)row * D_HID + col] = v;
      }
    }
  }
}

// -------------------------------------------------------------------------
// Kernel 1 (fallback, ws too small): f32 sources, convert during staging.
// -------------------------------------------------------------------------
__global__ __launch_bounds__(256, 2) void sae_encoder_wmma_f32(
    const float* __restrict__ x, const float* __restrict__ W_enc,
    const float* __restrict__ b_enc, float* __restrict__ code_out) {
  __shared__ u16 Asm[128 * 32];
  __shared__ u16 Bsm[128 * 32];

  const int tid  = threadIdx.x;
  const int lane = tid & 31;
  const int wave = tid >> 5;
  const int wm   = wave & 3;
  const int wn   = wave >> 2;
  const int rowBase = blockIdx.y * 128;
  const int colBase = blockIdx.x * 128;
  const int r16  = lane & 15;
  const int half = lane >> 4;

  v8f acc[2][4];
#pragma unroll
  for (int t = 0; t < 2; ++t)
#pragma unroll
    for (int cn = 0; cn < 4; ++cn)
      acc[t][cn] = (v8f){0.f, 0.f, 0.f, 0.f, 0.f, 0.f, 0.f, 0.f};

  for (int k0 = 0; k0 < D_IN; k0 += 32) {
#pragma unroll
    for (int i = 0; i < 4; ++i) {
      int q  = tid + i * 256;
      int r  = q >> 3;
      int kc = (q & 7) << 2;
      float4 av = *(const float4*)(x + (size_t)(rowBase + r) * D_IN + k0 + kc);
      float4 bv = *(const float4*)(W_enc + (size_t)(colBase + r) * D_IN + k0 + kc);
      uint2 ap, bp;
      ap.x = (unsigned)f2bf(av.x) | ((unsigned)f2bf(av.y) << 16);
      ap.y = (unsigned)f2bf(av.z) | ((unsigned)f2bf(av.w) << 16);
      bp.x = (unsigned)f2bf(bv.x) | ((unsigned)f2bf(bv.y) << 16);
      bp.y = (unsigned)f2bf(bv.z) | ((unsigned)f2bf(bv.w) << 16);
      *(uint2*)(&Asm[r * 32 + kc]) = ap;
      *(uint2*)(&Bsm[r * 32 + kc]) = bp;
    }
    if (k0 + 32 < D_IN) {
      int rn  = tid >> 3;
      int kcn = (tid & 7) << 2;
      __builtin_prefetch(x     + (size_t)(rowBase + rn) * D_IN + k0 + 32 + kcn, 0, 0);
      __builtin_prefetch(W_enc + (size_t)(colBase + rn) * D_IN + k0 + 32 + kcn, 0, 0);
    }
    __syncthreads();

    union U { v16bf v; unsigned u[8]; };
    U a[2], b[4];
#pragma unroll
    for (int t = 0; t < 2; ++t) {
      int row = wm * 32 + t * 16 + r16;
      int kbb = half * 8;
#pragma unroll
      for (int v = 0; v < 4; ++v)
        a[t].u[v] = *(const unsigned*)&Asm[row * 32 + kbb + 2 * v];
#pragma unroll
      for (int v = 4; v < 8; ++v)
        a[t].u[v] = *(const unsigned*)&Asm[row * 32 + kbb + 16 + 2 * (v - 4)];
    }
#pragma unroll
    for (int cn = 0; cn < 4; ++cn) {
      int col = wn * 64 + cn * 16 + r16;
#pragma unroll
      for (int g = 0; g < 8; ++g)
        b[cn].u[g] = *(const unsigned*)&Bsm[col * 32 + half * 8 + 2 * g + (g >= 4 ? 8 : 0)];
    }
#pragma unroll
    for (int t = 0; t < 2; ++t)
#pragma unroll
      for (int cn = 0; cn < 4; ++cn)
        acc[t][cn] = __builtin_amdgcn_wmma_f32_16x16x32_bf16(
            false, a[t].v, false, b[cn].v, (short)0, acc[t][cn], false, false);

    __syncthreads();
  }

#pragma unroll
  for (int t = 0; t < 2; ++t) {
#pragma unroll
    for (int cn = 0; cn < 4; ++cn) {
      int col = colBase + wn * 64 + cn * 16 + r16;
      float bias = b_enc[col];
#pragma unroll
      for (int g = 0; g < 8; ++g) {
        int row = rowBase + wm * 32 + t * 16 + half * 8 + g;
        float v = acc[t][cn][g] + bias;
        v = v > 0.f ? v : 0.f;
        code_out[(size_t)row * D_HID + col] = v;
      }
    }
  }
}

// -------------------------------------------------------------------------
// Kernel 2: per-row top-k(30) in place on code, then sparse decode -> recon
// -------------------------------------------------------------------------
__global__ __launch_bounds__(256, 1) void sae_topk_decode(
    const float* __restrict__ W_dec, float* __restrict__ recon,
    float* __restrict__ code) {
  __shared__ float rowS[D_HID];      // 64 KB (<< 320 KB/WGP)
  __shared__ float wval[8];
  __shared__ int   widx[8];
  __shared__ float selV[TOP_K];
  __shared__ int   selI[TOP_K];

  const int tid  = threadIdx.x;
  const int lane = tid & 31;
  const int wave = tid >> 5;
  const size_t row = blockIdx.x;
  float* crow = code + row * (size_t)D_HID;

  float4* rs4 = (float4*)rowS;
  float4* c4  = (float4*)crow;

#pragma unroll
  for (int i = 0; i < 16; ++i) {
    int q = tid + i * 256;
    rs4[q] = c4[q];
  }
  __syncthreads();

  const float NEG = -__builtin_huge_valf();

  for (int s = 0; s < TOP_K; ++s) {
    float lv = NEG; int li = -1;
    for (int i = 0; i < 64; ++i) {
      int idx = i * 256 + tid;
      float v = rowS[idx];
      if (v > lv) { lv = v; li = idx; }
    }
#pragma unroll
    for (int off = 16; off > 0; off >>= 1) {
      float ov = __shfl_down(lv, off);
      int   oi = __shfl_down(li, off);
      if (ov > lv) { lv = ov; li = oi; }
    }
    if (lane == 0) { wval[wave] = lv; widx[wave] = li; }
    __syncthreads();
    if (tid == 0) {
      float bv = wval[0]; int bi = widx[0];
      for (int w = 1; w < 8; ++w)
        if (wval[w] > bv) { bv = wval[w]; bi = widx[w]; }
      selV[s] = bv; selI[s] = bi;
      rowS[bi] = NEG;
    }
    __syncthreads();
  }

#pragma unroll
  for (int i = 0; i < 16; ++i) {
    int q = tid + i * 256;
    float4 v = rs4[q];
    float4 o;
    float* vp = (float*)&v;
    float* op = (float*)&o;
#pragma unroll
    for (int c = 0; c < 4; ++c) {
      float r = 0.f;
      if (vp[c] == NEG) {
        int gi = q * 4 + c;
        for (int s = 0; s < TOP_K; ++s)
          if (selI[s] == gi) r = selV[s];
      }
      op[c] = r;
    }
    c4[q] = o;
  }

#pragma unroll
  for (int ii = 0; ii < 3; ++ii) {
    int i = tid + ii * 256;
    float accv = 0.f;
    for (int s = 0; s < TOP_K; ++s)
      accv += selV[s] * W_dec[(size_t)i * D_HID + selI[s]];
    recon[row * D_IN + i] = accv;
  }
}

extern "C" void kernel_launch(void* const* d_in, const int* in_sizes, int n_in,
                              void* d_out, int out_size, void* d_ws, size_t ws_size,
                              hipStream_t stream) {
  const float* x     = (const float*)d_in[0];
  const float* W_enc = (const float*)d_in[1];
  const float* b_enc = (const float*)d_in[2];
  const float* W_dec = (const float*)d_in[3];

  float* out   = (float*)d_out;
  float* recon = out;                                  // [B, D_IN]
  float* code  = out + (size_t)B_DIM * D_IN;           // [B, D_HID]

  dim3 g1(D_HID / 128, B_DIM / 128);

  const size_t nx = (size_t)B_DIM * D_IN;
  const size_t nw = (size_t)D_HID * D_IN;
  const size_t need = (nx + nw) * sizeof(u16);

  if (ws_size >= need) {
    u16* xbf = (u16*)d_ws;
    u16* wbf = xbf + nx;
    f32_to_bf16_pack<<<dim3((unsigned)(nx / 8 / 256)), dim3(256), 0, stream>>>(
        x, xbf, (int)(nx / 8));
    f32_to_bf16_pack<<<dim3((unsigned)(nw / 8 / 256)), dim3(256), 0, stream>>>(
        W_enc, wbf, (int)(nw / 8));
    sae_encoder_wmma_bf16<<<g1, dim3(256), 0, stream>>>(xbf, wbf, b_enc, code);
  } else {
    sae_encoder_wmma_f32<<<g1, dim3(256), 0, stream>>>(x, W_enc, b_enc, code);
  }
  sae_topk_decode<<<dim3(B_DIM), dim3(256), 0, stream>>>(W_dec, recon, code);
}
